// Base_CNN_49735721287762
// MI455X (gfx1250) — matrix operand order repaired and checked by reading, not compiled, not run on hardware
//
#include <hip/hip_runtime.h>

// ---------------------------------------------------------------------------
// VSE triplet loss for MI455X (gfx1250, wave32, WMMA).
//   out[0]          = total_loss
//   out[1 .. 1+B*B) = sim matrix (row-major)
// Strategy: row-normalize + split fp32 -> bf16 hi/lo (bf16x3 GEMM trick),
// stored ROW-INTERLEAVED (row r: hi @ 2r*D, lo @ (2r+1)*D) so hi and lo are
// reachable from ONE per-lane pointer via immediate offsets. GEMM:
// v_wmma_f32_16x16x32_bf16, 64x64 tile per wave = 48 WMMA per k-step, with
// 8 loop-carried GLOBAL-addrspace pointers bumped +64B/iter, so every load
// is a global_load_b128 with a constant instruction offset (no FLAT, no
// DScnt coupling in the hot loop).
// Workspace: 2 interleaved u16 arrays (16 MB each, L2-resident) + diag.
// ---------------------------------------------------------------------------

typedef __attribute__((ext_vector_type(16))) __bf16 v16bf;
typedef __attribute__((ext_vector_type(8)))  __bf16 v8bf;
typedef __attribute__((ext_vector_type(8)))  float  v8f;

// Explicit global (addrspace 1) pointee types: guarantees global_load/store
// encodings instead of flat_* when pointers are loop-carried through PHIs.
#define GAS __attribute__((address_space(1)))
typedef GAS const unsigned short g_cu16;
typedef GAS const v16bf          g_cv16bf;
typedef GAS const v8bf           g_cv8bf;
typedef GAS float                g_f32;

#define BDIM 4096
#define DDIM 1024
#define MARGIN 0.2f

__device__ __forceinline__ unsigned short f32_to_bf16_rne(float v) {
    unsigned u = __float_as_uint(v);
    u += 0x7FFFu + ((u >> 16) & 1u);   // round-to-nearest-even
    return (unsigned short)(u >> 16);
}

// ---------------------------------------------------------------------------
// Kernel 1: L2-normalize each row, split into bf16 hi + lo, store interleaved.
// One 256-thread block per row; rows [0,B) = image, [B,2B) = text.
// ---------------------------------------------------------------------------
__global__ void __launch_bounds__(256)
normalize_split_kernel(const float* __restrict__ im, const float* __restrict__ tx,
                       unsigned short* __restrict__ Aint, unsigned short* __restrict__ Bint)
{
    const int row = blockIdx.x;
    const float* src;
    unsigned short* hi;
    if (row < BDIM) {
        src = im + (size_t)row * DDIM;
        hi  = Aint + (size_t)(2 * row) * DDIM;
    } else {
        const int r = row - BDIM;
        src = tx + (size_t)r * DDIM;
        hi  = Bint + (size_t)(2 * r) * DDIM;
    }
    unsigned short* lo = hi + DDIM;            // interleaved: lo row follows hi row

    const int t = threadIdx.x;                 // 256 threads * float4 = 1024 elems
    const float4 x = reinterpret_cast<const float4*>(src)[t];
    float ss = x.x * x.x + x.y * x.y + x.z * x.z + x.w * x.w;

    // wave32 shuffle reduction, then cross-wave via LDS
    #pragma unroll
    for (int off = 16; off; off >>= 1) ss += __shfl_down(ss, off, 32);
    __shared__ float wsum[8];
    if ((t & 31) == 0) wsum[t >> 5] = ss;
    __syncthreads();
    const float total = wsum[0] + wsum[1] + wsum[2] + wsum[3] +
                        wsum[4] + wsum[5] + wsum[6] + wsum[7];
    const float inv = 1.0f / fmaxf(sqrtf(total), 1e-12f);

    const float v[4] = { x.x * inv, x.y * inv, x.z * inv, x.w * inv };
    unsigned short hh[4], ll[4];
    #pragma unroll
    for (int i = 0; i < 4; ++i) {
        hh[i] = f32_to_bf16_rne(v[i]);
        const float hf = __uint_as_float((unsigned)hh[i] << 16);
        ll[i] = f32_to_bf16_rne(v[i] - hf);    // residual: v ~= hi + lo
    }
    ushort4 h = { hh[0], hh[1], hh[2], hh[3] };
    ushort4 l = { ll[0], ll[1], ll[2], ll[3] };
    reinterpret_cast<ushort4*>(hi)[t] = h;
    reinterpret_cast<ushort4*>(lo)[t] = l;
}

// ---------------------------------------------------------------------------
// Kernel 2: sim = A_norm @ B_norm^T via bf16x3 WMMA.
// Block = 256 threads = 8 wave32s arranged 2(M) x 4(N); wave tile = 64x64
// (4x4 fragments of 16x16), block tile = 128x256.
// ---------------------------------------------------------------------------
__device__ __forceinline__ v8f wmma_bf16(v16bf a, v16bf b, v8f c) {
    return __builtin_amdgcn_wmma_f32_16x16x32_bf16(
        /*neg_a=*/false, a, /*neg_b=*/false, b,
        /*c_mod=*/(short)0, c, /*reuse_a=*/false, /*reuse_b=*/false);
}

__global__ void __launch_bounds__(256)
sim_gemm_kernel(const unsigned short* __restrict__ Aint,
                const unsigned short* __restrict__ Bint,
                float* __restrict__ sim)
{
    const int lane  = threadIdx.x & 31;
    const int wave  = threadIdx.x >> 5;
    const int lr    = lane & 15;     // row/col within fragment
    const int lh    = lane >> 4;     // lane half selects K sub-range
    const int waveM = wave & 1;      // 2 waves along M
    const int waveN = wave >> 1;     // 4 waves along N
    const int mWave = blockIdx.y * 128 + waveM * 64;
    const int nWave = blockIdx.x * 256 + waveN * 64;

    // 8 loop-carried per-lane GLOBAL pointers. From each, every operand of one
    // k-step is a constant immediate offset:
    //   A (16x32, contiguous K order, not the ISA table): lanes 0-15 hold K [0..16), lanes 16-31
    //   K [16..32): hi @ +0,+16B ; lo @ +2048,+2064B (interleaved row).
    //   B (32x16 layout): lane n holds 16 contiguous K: hi @ +0 (32B) ; lo @ +2048.
    g_cu16* pA[4];
    g_cu16* pB[4];
    #pragma unroll
    for (int mi = 0; mi < 4; ++mi)
        pA[mi] = (g_cu16*)(Aint + (size_t)(2 * (mWave + mi * 16 + lr)) * DDIM + 16 * lh);
    #pragma unroll
    for (int ni = 0; ni < 4; ++ni)
        pB[ni] = (g_cu16*)(Bint + (size_t)(2 * (nWave + ni * 16 + lr)) * DDIM + 16 * lh);

    v8f acc[4][4] = {};

    for (int k0 = 0; k0 < DDIM; k0 += 32) {
        // Stage all B fragments for this k-step (shared by every mi).
        v16bf bHi[4], bLo[4];
        #pragma unroll
        for (int ni = 0; ni < 4; ++ni) {
            bHi[ni] = *(g_cv16bf*)(pB[ni]);
            bLo[ni] = *(g_cv16bf*)(pB[ni] + DDIM);
        }

        // One mi at a time keeps only one A hi/lo pair live.
        #pragma unroll
        for (int mi = 0; mi < 4; ++mi) {
            const v8bf h0 = *(g_cv8bf*)(pA[mi]);
            const v8bf h1 = *(g_cv8bf*)(pA[mi] + 8);
            const v8bf l0 = *(g_cv8bf*)(pA[mi] + DDIM);
            const v8bf l1 = *(g_cv8bf*)(pA[mi] + DDIM + 8);
            const v16bf aHi = __builtin_shufflevector(h0, h1, 0,1,2,3,4,5,6,7,8,9,10,11,12,13,14,15);
            const v16bf aLo = __builtin_shufflevector(l0, l1, 0,1,2,3,4,5,6,7,8,9,10,11,12,13,14,15);
            #pragma unroll
            for (int ni = 0; ni < 4; ++ni) {
                // bf16x3: hi*hi + hi*lo + lo*hi, f32 accumulate (~fp32 exact)
                acc[mi][ni] = wmma_bf16(aHi, bHi[ni], acc[mi][ni]);
                acc[mi][ni] = wmma_bf16(aHi, bLo[ni], acc[mi][ni]);
                acc[mi][ni] = wmma_bf16(aLo, bHi[ni], acc[mi][ni]);
            }
        }

        // Advance all pointers by 32 K-elements (64 bytes): 8 v_add_nc_u64.
        #pragma unroll
        for (int mi = 0; mi < 4; ++mi) pA[mi] += 32;
        #pragma unroll
        for (int ni = 0; ni < 4; ++ni) pB[ni] += 32;
    }

    // C layout: VGPR g -> row g (lanes 0-15) / row g+8 (lanes 16-31), col = lane&15.
    // Per-tile global base pointer; rows become g*16384-byte immediate offsets.
    #pragma unroll
    for (int mi = 0; mi < 4; ++mi)
        #pragma unroll
        for (int ni = 0; ni < 4; ++ni) {
            g_f32* p = (g_f32*)(sim + (size_t)(mWave + mi * 16 + 8 * lh) * BDIM
                                    + (nWave + ni * 16 + lr));
            #pragma unroll
            for (int g = 0; g < 8; ++g)
                p[(size_t)g * BDIM] = acc[mi][ni][g];
        }
}

// ---------------------------------------------------------------------------
// Kernel 3: pull out the diagonal (positive-pair scores).
// ---------------------------------------------------------------------------
__global__ void __launch_bounds__(256)
diag_extract_kernel(const float* __restrict__ sim, float* __restrict__ diag)
{
    const int i = blockIdx.x * 256 + threadIdx.x;
    if (i < BDIM) diag[i] = sim[(size_t)i * BDIM + i];
}

// ---------------------------------------------------------------------------
// Kernel 4: total_loss = sum_{i!=j} relu(M+s_ij-s_ii) + relu(M+s_ij-s_jj)
// ---------------------------------------------------------------------------
__global__ void __launch_bounds__(256)
loss_reduce_kernel(const float* __restrict__ sim, const float* __restrict__ diag,
                   float* __restrict__ out0)
{
    const long long NN = (long long)BDIM * BDIM;
    const long long stride = (long long)gridDim.x * blockDim.x;
    float local = 0.f;
    for (long long idx = (long long)blockIdx.x * blockDim.x + threadIdx.x;
         idx < NN; idx += stride) {
        const int i = (int)(idx >> 12);
        const int j = (int)(idx & (BDIM - 1));
        const float s = sim[idx];
        const float c = fmaxf(MARGIN + s - diag[i], 0.f) +
                        fmaxf(MARGIN + s - diag[j], 0.f);
        local += (i == j) ? 0.f : c;
    }
    #pragma unroll
    for (int off = 16; off; off >>= 1) local += __shfl_down(local, off, 32);
    __shared__ float wsum[8];
    const int lane = threadIdx.x & 31, w = threadIdx.x >> 5;
    if (lane == 0) wsum[w] = local;
    __syncthreads();
    if (w == 0) {
        float v = (lane < 8) ? wsum[lane] : 0.f;
        #pragma unroll
        for (int off = 4; off; off >>= 1) v += __shfl_down(v, off, 32);
        if (lane == 0) atomicAdd(out0, v);
    }
}

// ---------------------------------------------------------------------------
extern "C" void kernel_launch(void* const* d_in, const int* in_sizes, int n_in,
                              void* d_out, int out_size, void* d_ws, size_t ws_size,
                              hipStream_t stream)
{
    const float* im = (const float*)d_in[0];   // [4096, 1024] fp32
    const float* tx = (const float*)d_in[1];   // [4096, 1024] fp32
    float* out = (float*)d_out;                // [0]=loss, [1..]=sim
    float* sim = out + 1;

    const size_t elems = (size_t)BDIM * DDIM;
    unsigned short* Aint = (unsigned short*)d_ws;     // interleaved hi/lo, 16 MB
    unsigned short* Bint = Aint + 2 * elems;          // interleaved hi/lo, 16 MB
    float* diag = (float*)(Bint + 2 * elems);         // 16 KB

    // 1) normalize + bf16 hi/lo split (2*B rows)
    normalize_split_kernel<<<2 * BDIM, 256, 0, stream>>>(im, tx, Aint, Bint);

    // 2) WMMA GEMM: block tile 128x256, grid (4096/256, 4096/128)
    sim_gemm_kernel<<<dim3(BDIM / 256, BDIM / 128), 256, 0, stream>>>(Aint, Bint, sim);

    // 3) diagonal
    diag_extract_kernel<<<BDIM / 256, 256, 0, stream>>>(sim, diag);

    // 4) hinge-loss reduction (zero the scalar first; memset is graph-capturable)
    hipMemsetAsync(d_out, 0, sizeof(float), stream);
    loss_reduce_kernel<<<2048, 256, 0, stream>>>(sim, diag, out);
}
